// Generator_57964878627091
// MI455X (gfx1250) — matrix operand order repaired and checked by reading, not compiled, not run on hardware
//
#include <hip/hip_runtime.h>
#include <hip/hip_bf16.h>

// ---------------------------------------------------------------------------
// MI455X (gfx1250): wave32, WMMA f32_16x16x32_f16 for all Gram/GEMM work,
// async global->LDS staging (ASYNCcnt) for block-shared operands.
// ---------------------------------------------------------------------------

typedef _Float16 v16h __attribute__((ext_vector_type(16)));
typedef _Float16 v8h  __attribute__((ext_vector_type(8)));
typedef float    v8f  __attribute__((ext_vector_type(8)));

#define NPTS  2048
#define KNN   8
#define FDIM  128
#define NBLK  12

// A (16x32 f16, MxK) loader, contiguous k order shared with B loads (wave32):
// lane L<16 holds row M=L, K = kb..kb+15;
// lane L>=16 holds row M=L-16, K = kb+16..kb+31.
__device__ __forceinline__ v16h load_a16(const _Float16* base, int stride,
                                         int lane, int kb) {
  const _Float16* p = base + (size_t)(lane & 15) * stride + kb + ((lane >= 16) ? 16 : 0);
  v8h lo = *(const v8h*)(p);
  v8h hi = *(const v8h*)(p + 8);
  v16h a;
#pragma unroll
  for (int e = 0; e < 8; ++e) { a[e] = lo[e]; a[e + 8] = hi[e]; }
  return a;
}

// ---- KNN: top-8 neighbours via WMMA Gram tiles ----------------------------
// grid.x = nq/128, block = 256 (8 waves, each owns 16 query rows).
// Column tiles are staged block-wide into LDS with async loads, double
// buffered: stage ct+16 while computing ct.
template <int KCHUNKS>
__global__ void __launch_bounds__(256)
knn_kernel(const _Float16* __restrict__ x, const float* __restrict__ nrm,
           int nc, int* __restrict__ nbr) {
  constexpr int DPAD = KCHUNKS * 32;
  constexpr int CH   = (16 * DPAD) / 8;          // 16-byte chunks per tile
  __shared__ alignas(32) _Float16 colbuf[2][16 * DPAD];
  __shared__ float dtile[8][16][16];
  const int tid   = threadIdx.x;
  const int lane  = tid & 31;
  const int wave  = tid >> 5;
  const int qbase = (blockIdx.x * 8 + wave) * 16;
  const int nn    = lane & 15;
  const int mb    = (lane >= 16) ? 8 : 0;

  // Query tile A held in registers for the whole column sweep.
  const _Float16* abase = x + (size_t)qbase * DPAD;
  v16h areg[KCHUNKS];
#pragma unroll
  for (int kc = 0; kc < KCHUNKS; ++kc) areg[kc] = load_a16(abase, DPAD, lane, kc * 32);

  float qn8[8];
#pragma unroll
  for (int r = 0; r < 8; ++r) qn8[r] = nrm[qbase + r + mb];

  float bestd[KNN]; int besti[KNN];
#pragma unroll
  for (int j = 0; j < KNN; ++j) { bestd[j] = 3.0e38f; besti[j] = 0; }

  // prologue: async-stage column tile 0
  if (tid < CH) {
    unsigned lo = (unsigned)(size_t)&colbuf[0][tid * 8];
    const _Float16* g = x + (size_t)tid * 8;
    asm volatile("global_load_async_to_lds_b128 %0, %1, off"
                 :: "v"(lo), "v"(g) : "memory");
  }
  asm volatile("s_wait_asynccnt 0" ::: "memory");
  __syncthreads();

  int buf = 0;
  for (int ct = 0; ct < nc; ct += 16, buf ^= 1) {
    // stage next column tile while we compute on the current one
    if (ct + 16 < nc && tid < CH) {
      unsigned lo = (unsigned)(size_t)&colbuf[buf ^ 1][tid * 8];
      const _Float16* g = x + (size_t)(ct + 16) * DPAD + tid * 8;
      asm volatile("global_load_async_to_lds_b128 %0, %1, off"
                   :: "v"(lo), "v"(g) : "memory");
    }
    v16h breg[KCHUNKS];
#pragma unroll
    for (int kc = 0; kc < KCHUNKS; ++kc)
      breg[kc] = *(const v16h*)&colbuf[buf][(size_t)nn * DPAD + kc * 32 +
                                            ((lane >= 16) ? 16 : 0)];
    v8f acc = {};
#pragma unroll
    for (int kc = 0; kc < KCHUNKS; ++kc)
      acc = __builtin_amdgcn_wmma_f32_16x16x32_f16(
          false, areg[kc], false, breg[kc], (short)0, acc, false, false);

    const float cn = nrm[ct + nn];
#pragma unroll
    for (int r = 0; r < 8; ++r) {
      float d2 = qn8[r] + cn - 2.0f * acc[r];
      if (ct + nn == qbase + r + mb) d2 = 3.0e38f;   // exclude self loop
      dtile[wave][r + mb][nn] = d2;
    }
    asm volatile("s_wait_dscnt 0" ::: "memory");     // in-wave LDS RAW
    if (lane < 16) {
#pragma unroll
      for (int c = 0; c < 16; ++c) {
        float d2 = dtile[wave][lane][c];
        if (d2 < bestd[KNN - 1]) {
          bestd[KNN - 1] = d2; besti[KNN - 1] = ct + c;
#pragma unroll
          for (int j = KNN - 2; j >= 0; --j) {
            if (bestd[j + 1] < bestd[j]) {
              float td = bestd[j]; bestd[j] = bestd[j + 1]; bestd[j + 1] = td;
              int ti = besti[j]; besti[j] = besti[j + 1]; besti[j + 1] = ti;
            }
          }
        }
      }
    }
    asm volatile("s_wait_asynccnt 0" ::: "memory");  // next tile landed
    __syncthreads();                                 // all waves done with buf
  }
  if (lane < 16) {
#pragma unroll
    for (int j = 0; j < KNN; ++j)
      nbr[(size_t)(qbase + lane) * KNN + j] = besti[j];
  }
}

// ---- GEMM: C(n x 128) = A(n x DPAD) * Wt(128 x DPAD)^T --------------------
// Wt is block-shared: async-stage it into LDS once, all waves read from LDS.
template <int KCHUNKS>
__global__ void __launch_bounds__(256)
gemm_kernel(const _Float16* __restrict__ A, const _Float16* __restrict__ Wt,
            float* __restrict__ C) {
  constexpr int DPAD = KCHUNKS * 32;
  constexpr int CH   = (128 * DPAD) / 8;
  __shared__ alignas(32) _Float16 wlds[128 * DPAD];
  const int tid   = threadIdx.x;
  const int lane  = tid & 31;
  const int wave  = tid >> 5;
  const int rbase = (blockIdx.x * 8 + wave) * 16;
  const int nn    = lane & 15;
  const int mb    = (lane >= 16) ? 8 : 0;

  for (int ch = tid; ch < CH; ch += 256) {
    unsigned lo = (unsigned)(size_t)&wlds[ch * 8];
    const _Float16* g = Wt + (size_t)ch * 8;
    asm volatile("global_load_async_to_lds_b128 %0, %1, off"
                 :: "v"(lo), "v"(g) : "memory");
  }
  const _Float16* abase = A + (size_t)rbase * DPAD;
  v16h areg[KCHUNKS];
#pragma unroll
  for (int kc = 0; kc < KCHUNKS; ++kc) areg[kc] = load_a16(abase, DPAD, lane, kc * 32);
  asm volatile("s_wait_asynccnt 0" ::: "memory");
  __syncthreads();

#pragma unroll
  for (int ct = 0; ct < 128; ct += 16) {
    v16h breg[KCHUNKS];
#pragma unroll
    for (int kc = 0; kc < KCHUNKS; ++kc)
      breg[kc] = *(const v16h*)&wlds[(size_t)(ct + nn) * DPAD + kc * 32 +
                                     ((lane >= 16) ? 16 : 0)];
    v8f acc = {};
#pragma unroll
    for (int kc = 0; kc < KCHUNKS; ++kc)
      acc = __builtin_amdgcn_wmma_f32_16x16x32_f16(
          false, areg[kc], false, breg[kc], (short)0, acc, false, false);
#pragma unroll
    for (int r = 0; r < 8; ++r)
      C[(size_t)(rbase + r + mb) * 128 + ct + nn] = acc[r];
  }
}

// ---- elementwise / gather kernels -----------------------------------------
__global__ void xyz2pos_kernel(const float* __restrict__ xyz, float* __restrict__ pos, int n) {
  int idx = blockIdx.x * blockDim.x + threadIdx.x;
  if (idx >= n * 3) return;
  int i = idx / 3, c = idx % 3;
  pos[idx] = xyz[(size_t)c * n + i];
}

__global__ void cvt16_kernel(const float* __restrict__ s, _Float16* __restrict__ d,
                             int n, int dsrc, int dpad) {
  int idx = blockIdx.x * blockDim.x + threadIdx.x;
  if (idx >= n * dpad) return;
  int i = idx / dpad, c = idx % dpad;
  d[idx] = (c < dsrc) ? (_Float16)s[(size_t)i * dsrc + c] : (_Float16)0.f;
}

// Wt[ncol][k] = W[k][ncol]  (K-major, zero padded to dpad)
__global__ void cvtwt_kernel(const float* __restrict__ W, _Float16* __restrict__ Wt,
                             int d, int ncols, int dpad) {
  int idx = blockIdx.x * blockDim.x + threadIdx.x;
  if (idx >= ncols * dpad) return;
  int nidx = idx / dpad, kk = idx % dpad;
  Wt[idx] = (kk < d) ? (_Float16)W[(size_t)kk * ncols + nidx] : (_Float16)0.f;
}

__global__ void norms_kernel(const float* __restrict__ x, float* __restrict__ o, int n, int d) {
  int i = blockIdx.x * blockDim.x + threadIdx.x;
  if (i >= n) return;
  float s = 0.f;
  for (int c = 0; c < d; ++c) { float v = x[(size_t)i * d + c]; s += v * v; }
  o[i] = s;
}

// agg = (sum_nbr hw + hw)/(k+1) + bias ; optional relu ; optional residual
__global__ void agg_kernel(const float* __restrict__ hw, const int* __restrict__ nbr,
                           const float* __restrict__ bias, const float* res,
                           float* out, _Float16* outh, int n, int dorelu) {
  int idx = blockIdx.x * blockDim.x + threadIdx.x;
  if (idx >= n * FDIM) return;
  int i = idx >> 7, c = idx & 127;
  float s = hw[idx];
#pragma unroll
  for (int j = 0; j < KNN; ++j)
    s += hw[(size_t)nbr[(size_t)i * KNN + j] * FDIM + c];
  s = s * (1.0f / (KNN + 1.0f)) + bias[c];
  if (dorelu) s = fmaxf(s, 0.f);
  if (res) s += res[idx];
  out[idx] = s;
  if (outh) outh[idx] = (_Float16)s;
}

__global__ void fill0_kernel(float* p, int tot) {
  int idx = blockIdx.x * blockDim.x + threadIdx.x;
  if (idx < tot) p[idx] = 0.f;
}

// scatter-max over edges; features are post-relu >= 0 so int-bit atomicMax is
// order-preserving and the 0-init matches torch_scatter's empty-row = 0.
__global__ void segmax_kernel(const float* __restrict__ h, const int* __restrict__ nbr,
                              float* out, int n) {
  int idx = blockIdx.x * blockDim.x + threadIdx.x;
  if (idx >= n * FDIM) return;
  int i = idx >> 7, c = idx & 127;
  int vi = __float_as_int(h[idx]);
#pragma unroll
  for (int j = 0; j < KNN; ++j) {
    int t = nbr[(size_t)i * KNN + j];
    atomicMax((int*)&out[(size_t)t * FDIM + c], vi);
  }
}

__global__ void ln_kernel(const float* __restrict__ y, const float* __restrict__ b0,
                          const float* __restrict__ g, const float* __restrict__ bb,
                          float* __restrict__ z, int n) {
  int i = blockIdx.x * blockDim.x + threadIdx.x;
  if (i >= n) return;
  const float* row = y + (size_t)i * FDIM;
  float mu = 0.f;
  for (int c = 0; c < FDIM; ++c) mu += row[c] + b0[c];
  mu *= (1.0f / FDIM);
  float var = 0.f;
  for (int c = 0; c < FDIM; ++c) { float t = row[c] + b0[c] - mu; var += t * t; }
  var *= (1.0f / FDIM);
  float inv = rsqrtf(var + 1e-5f);
  for (int c = 0; c < FDIM; ++c) {
    float t = (row[c] + b0[c] - mu) * inv * g[c] + bb[c];
    z[(size_t)i * FDIM + c] = fmaxf(t, 0.f);
  }
}

// off = tanh(z @ oW2 + ob2)*0.1 ; new points row-major + optional (3,2n) out
__global__ void offset_kernel(const float* __restrict__ z, const float* __restrict__ oW2,
                              const float* __restrict__ ob2, const float* __restrict__ pos_in,
                              float* __restrict__ pos_out, float* out_cm, int n) {
  int i = blockIdx.x * blockDim.x + threadIdx.x;
  if (i >= n) return;
  float off[6];
#pragma unroll
  for (int t = 0; t < 6; ++t) off[t] = ob2[t];
  const float* zr = z + (size_t)i * FDIM;
  for (int c = 0; c < FDIM; ++c) {
    float zv = zr[c];
#pragma unroll
    for (int t = 0; t < 6; ++t) off[t] += zv * oW2[c * 6 + t];
  }
#pragma unroll
  for (int t = 0; t < 6; ++t) off[t] = tanhf(off[t]) * 0.1f;
#pragma unroll
  for (int c = 0; c < 3; ++c)
#pragma unroll
    for (int s = 0; s < 2; ++s) {
      float v = pos_in[(size_t)i * 3 + c] + off[c * 2 + s];
      pos_out[(size_t)((size_t)s * n + i) * 3 + c] = v;
      if (out_cm) out_cm[(size_t)c * (2 * (size_t)n) + (size_t)s * n + i] = v;
    }
}

// inverse-distance weighted interpolation over valid (old-set) neighbours
__global__ void interp_kernel(const float* __restrict__ feat, const float* __restrict__ pn,
                              const float* __restrict__ po, const int* __restrict__ nbr,
                              float* __restrict__ fi, int m) {
  int idx = blockIdx.x * blockDim.x + threadIdx.x;
  if (idx >= m * FDIM) return;
  int q = idx >> 7, c = idx & 127;
  float qx = pn[(size_t)q * 3], qy = pn[(size_t)q * 3 + 1], qz = pn[(size_t)q * 3 + 2];
  float num = 0.f, den = 0.f;
#pragma unroll
  for (int j = 0; j < KNN; ++j) {
    int nb = nbr[(size_t)q * KNN + j];
    if (nb >= m) {
      int oi = nb - m;
      float dx = qx - po[(size_t)oi * 3];
      float dy = qy - po[(size_t)oi * 3 + 1];
      float dz = qz - po[(size_t)oi * 3 + 2];
      float inv = 1.0f / (sqrtf(dx * dx + dy * dy + dz * dz) + 1e-5f);
      den += inv;
      num += inv * feat[(size_t)oi * FDIM + c];
    }
  }
  fi[idx] = num / fmaxf(den, 1e-12f);
}

// ---------------------------------------------------------------------------
extern "C" void kernel_launch(void* const* d_in, const int* in_sizes, int n_in,
                              void* d_out, int out_size, void* d_ws, size_t ws_size,
                              hipStream_t stream) {
  (void)in_sizes; (void)n_in; (void)out_size; (void)ws_size;
  const float* xyz   = (const float*)d_in[0];
  const float* fW[3] = {(const float*)d_in[1], (const float*)d_in[3], (const float*)d_in[5]};
  const float* fb[3] = {(const float*)d_in[2], (const float*)d_in[4], (const float*)d_in[6]};
  struct UP { const float *W1,*b1,*W2,*b2,*oW1,*ob1,*lng,*lnb,*oW2,*ob2; } up[2];
  for (int u = 0; u < 2; ++u) {
    int o = 7 + u * 10;
    up[u].W1  = (const float*)d_in[o + 0]; up[u].b1  = (const float*)d_in[o + 1];
    up[u].W2  = (const float*)d_in[o + 2]; up[u].b2  = (const float*)d_in[o + 3];
    up[u].oW1 = (const float*)d_in[o + 4]; up[u].ob1 = (const float*)d_in[o + 5];
    up[u].lng = (const float*)d_in[o + 6]; up[u].lnb = (const float*)d_in[o + 7];
    up[u].oW2 = (const float*)d_in[o + 8]; up[u].ob2 = (const float*)d_in[o + 9];
  }

  const int N1 = NPTS, N2 = 2 * NPTS, NPC = 3 * NPTS;   // 2048, 4096, 6144

  char* wsp = (char*)d_ws;
  auto alloc = [&](size_t bytes) {
    char* p = wsp; wsp += (bytes + 255) & ~(size_t)255; return (void*)p;
  };
  float*    pos0   = (float*)alloc((size_t)N1 * 3 * 4);
  float*    posA   = (float*)alloc((size_t)N2 * 3 * 4);
  float*    posB   = (float*)alloc((size_t)2 * N2 * 3 * 4);
  float*    pc     = (float*)alloc((size_t)NPC * 3 * 4);
  float*    hA     = (float*)alloc((size_t)N2 * FDIM * 4);
  float*    hB     = (float*)alloc((size_t)N2 * FDIM * 4);
  float*    hw     = (float*)alloc((size_t)N2 * FDIM * 4);
  float*    zbuf   = (float*)alloc((size_t)N2 * FDIM * 4);
  float*    featk  = (float*)alloc((size_t)N1 * FDIM * 4);
  _Float16* hf16   = (_Float16*)alloc((size_t)N2 * FDIM * 2);
  _Float16* h1f16  = (_Float16*)alloc((size_t)N2 * FDIM * 2);
  _Float16* pcf16  = (_Float16*)alloc((size_t)NPC * 32 * 2);
  _Float16* wt     = (_Float16*)alloc((size_t)FDIM * FDIM * 2);
  float*    nrm    = (float*)alloc((size_t)NPC * 4);
  int*      nbrb   = (int*)alloc((size_t)NPC * KNN * 4);

  auto ew = [](int tot) { return dim3((unsigned)((tot + 255) / 256)); };

  auto run_unpool = [&](int n, const UP& P, const float* pos_in, float* pos_out,
                        float* out_cm) {
    // entry invariant: hB holds features, hf16 = f16(hB)
    for (int blk = 0; blk < NBLK; ++blk) {
      norms_kernel<<<ew(n), 256, 0, stream>>>(hB, nrm, n, FDIM);
      knn_kernel<4><<<dim3((unsigned)(n / 128)), 256, 0, stream>>>(hf16, nrm, n, nbrb);
      cvtwt_kernel<<<ew(FDIM * FDIM), 256, 0, stream>>>(P.W1 + (size_t)blk * FDIM * FDIM, wt, FDIM, FDIM, FDIM);
      gemm_kernel<4><<<dim3((unsigned)(n / 128)), 256, 0, stream>>>(hf16, wt, hw);
      agg_kernel<<<ew(n * FDIM), 256, 0, stream>>>(hw, nbrb, P.b1 + (size_t)blk * FDIM,
                                                   nullptr, hA, h1f16, n, 1);
      cvtwt_kernel<<<ew(FDIM * FDIM), 256, 0, stream>>>(P.W2 + (size_t)blk * FDIM * FDIM, wt, FDIM, FDIM, FDIM);
      gemm_kernel<4><<<dim3((unsigned)(n / 128)), 256, 0, stream>>>(h1f16, wt, hw);
      agg_kernel<<<ew(n * FDIM), 256, 0, stream>>>(hw, nbrb, P.b2 + (size_t)blk * FDIM,
                                                   hB, hB, hf16, n, 0);   // + residual
    }
    cvtwt_kernel<<<ew(FDIM * FDIM), 256, 0, stream>>>(P.oW1, wt, FDIM, FDIM, FDIM);
    gemm_kernel<4><<<dim3((unsigned)(n / 128)), 256, 0, stream>>>(hf16, wt, hw);
    ln_kernel<<<ew(n), 256, 0, stream>>>(hw, P.ob1, P.lng, P.lnb, zbuf, n);
    offset_kernel<<<ew(n), 256, 0, stream>>>(zbuf, P.oW2, P.ob2, pos_in, pos_out, out_cm, n);
  };

  for (int b = 0; b < 2; ++b) {
    float* outb = (float*)d_out + (size_t)b * 3 * (2 * N2);

    // ---------------- featurenet (n = 2048) ----------------
    xyz2pos_kernel<<<ew(N1 * 3), 256, 0, stream>>>(xyz + (size_t)b * 3 * N1, pos0, N1);
    cvt16_kernel<<<ew(N1 * 32), 256, 0, stream>>>(pos0, pcf16, N1, 3, 32);
    norms_kernel<<<ew(N1), 256, 0, stream>>>(pos0, nrm, N1, 3);
    knn_kernel<1><<<dim3((unsigned)(N1 / 128)), 256, 0, stream>>>(pcf16, nrm, N1, nbrb);
    cvtwt_kernel<<<ew(FDIM * 32), 256, 0, stream>>>(fW[0], wt, 3, FDIM, 32);
    gemm_kernel<1><<<dim3((unsigned)(N1 / 128)), 256, 0, stream>>>(pcf16, wt, hw);
    agg_kernel<<<ew(N1 * FDIM), 256, 0, stream>>>(hw, nbrb, fb[0], nullptr, hA, nullptr, N1, 1);
    fill0_kernel<<<ew(N1 * FDIM), 256, 0, stream>>>(hB, N1 * FDIM);
    segmax_kernel<<<ew(N1 * FDIM), 256, 0, stream>>>(hA, nbrb, hB, N1);
    for (int L = 1; L < 3; ++L) {
      cvt16_kernel<<<ew(N1 * FDIM), 256, 0, stream>>>(hB, hf16, N1, FDIM, FDIM);
      norms_kernel<<<ew(N1), 256, 0, stream>>>(hB, nrm, N1, FDIM);
      knn_kernel<4><<<dim3((unsigned)(N1 / 128)), 256, 0, stream>>>(hf16, nrm, N1, nbrb);
      cvtwt_kernel<<<ew(FDIM * FDIM), 256, 0, stream>>>(fW[L], wt, FDIM, FDIM, FDIM);
      gemm_kernel<4><<<dim3((unsigned)(N1 / 128)), 256, 0, stream>>>(hf16, wt, hw);
      agg_kernel<<<ew(N1 * FDIM), 256, 0, stream>>>(hw, nbrb, fb[L], nullptr, hA, nullptr, N1, 1);
      fill0_kernel<<<ew(N1 * FDIM), 256, 0, stream>>>(hB, N1 * FDIM);
      segmax_kernel<<<ew(N1 * FDIM), 256, 0, stream>>>(hA, nbrb, hB, N1);
    }
    hipMemcpyAsync(featk, hB, (size_t)N1 * FDIM * 4, hipMemcpyDeviceToDevice, stream);

    // ---------------- unpool 1 (2048 -> 4096 points) ----------------
    cvt16_kernel<<<ew(N1 * FDIM), 256, 0, stream>>>(hB, hf16, N1, FDIM, FDIM);
    run_unpool(N1, up[0], pos0, posA, nullptr);

    // ---------------- interpolate onto new points ----------------
    hipMemcpyAsync(pc, posA, (size_t)N2 * 3 * 4, hipMemcpyDeviceToDevice, stream);
    hipMemcpyAsync(pc + (size_t)N2 * 3, pos0, (size_t)N1 * 3 * 4,
                   hipMemcpyDeviceToDevice, stream);
    cvt16_kernel<<<ew(NPC * 32), 256, 0, stream>>>(pc, pcf16, NPC, 3, 32);
    norms_kernel<<<ew(NPC), 256, 0, stream>>>(pc, nrm, NPC, 3);
    knn_kernel<1><<<dim3((unsigned)(N2 / 128)), 256, 0, stream>>>(pcf16, nrm, NPC, nbrb);
    interp_kernel<<<ew(N2 * FDIM), 256, 0, stream>>>(featk, posA, pos0, nbrb, hB, N2);

    // ---------------- unpool 2 (4096 -> 8192 points, final output) --------
    cvt16_kernel<<<ew(N2 * FDIM), 256, 0, stream>>>(hB, hf16, N2, FDIM, FDIM);
    run_unpool(N2, up[1], posA, posB, outb);
  }
}
